// DeepseekSparseAttention_18193481465899
// MI455X (gfx1250) — matrix operand order repaired and checked by reading, not compiled, not run on hardware
//
#include <hip/hip_runtime.h>
#include <hip/hip_bf16.h>
#include <math.h>

typedef __bf16 bf16_t;
typedef __attribute__((ext_vector_type(16))) __bf16 v16bf;
typedef __attribute__((ext_vector_type(8)))  __bf16 v8bf;
typedef __attribute__((ext_vector_type(4)))  __bf16 v4bf;
typedef __attribute__((ext_vector_type(4)))  float   v4f;
typedef __attribute__((ext_vector_type(8)))  float   v8f;

#define S_LEN 2048
#define DM    2048
#define HN    16
#define HDIM  128
#define RDIM  64
#define QLRN  1024
#define TOPK_N 512

#define WAIT_ASYNC24() asm volatile("s_wait_asynccnt 0x18" ::: "memory")
#define WAIT_ASYNC0()  asm volatile("s_wait_asynccnt 0x0"  ::: "memory")
#define WAIT_DS0()     asm volatile("s_wait_dscnt 0x0"     ::: "memory")

__device__ __forceinline__ bf16_t f2bf(float f){
  unsigned u = __float_as_uint(f);
  unsigned r = (u + 0x7FFFu + ((u >> 16) & 1u)) >> 16;
  union { unsigned short s; bf16_t b; } cv; cv.s = (unsigned short)r; return cv.b;
}

// ---------------- f32 -> bf16 convert (4 elems / lane) ----------------
__global__ void k_cvt_bf16(const float* __restrict__ src, bf16_t* __restrict__ dst, int n4){
  int i = blockIdx.x * 256 + threadIdx.x;
  if (i < n4){
    v4f f = *(const v4f*)(src + (size_t)i * 4);
    v4bf b;
#pragma unroll
    for (int j = 0; j < 4; j++) b[j] = f2bf(f[j]);
    *(v4bf*)(dst + (size_t)i * 4) = b;
  }
}

// ---------------- simple WMMA GEMM (one wave per 16x16 tile), used for N=16 ----------------
__global__ void k_gemm_bf16(const bf16_t* __restrict__ A, const bf16_t* __restrict__ W,
                            const float* __restrict__ bias, float* __restrict__ C,
                            int N, int K){
  int tn = blockIdx.x, tm = blockIdx.y;
  int lane = threadIdx.x, half = lane >> 4, l16 = lane & 15;
  const bf16_t* arow = A + (size_t)(tm * 16 + l16) * K;
  const bf16_t* wrow = W + (size_t)(tn * 16 + l16) * K;
  v8f c = {};
  for (int k0 = 0; k0 < K; k0 += 32){
    __builtin_prefetch(arow + k0 + 256, 0, 0);
    v8bf a0 = *(const v8bf*)(arow + k0 + half * 8);
    v8bf a1 = *(const v8bf*)(arow + k0 + 16 + half * 8);
    v8bf b0 = *(const v8bf*)(wrow + k0 + half * 8);
    v8bf b1 = *(const v8bf*)(wrow + k0 + 16 + half * 8);
    v16bf av, bv;
#pragma unroll
    for (int i = 0; i < 8; i++){ av[i]=a0[i]; av[8+i]=a1[i]; bv[i]=b0[i]; bv[8+i]=b1[i]; }
    c = __builtin_amdgcn_wmma_f32_16x16x32_bf16(false, av, false, bv, (short)0, c, false, false);
  }
  float b = bias[tn * 16 + l16];
#pragma unroll
  for (int v = 0; v < 8; v++)
    C[(size_t)(tm * 16 + v + half * 8) * N + tn * 16 + l16] = c[v] + b;
}

// ---------------- register-blocked WMMA GEMM: 32x64 per wave (2 M-tiles x 4 N-tiles) ----------------
__global__ void k_gemm_bf16_big(const bf16_t* __restrict__ A, const bf16_t* __restrict__ W,
                                const float* __restrict__ bias, float* __restrict__ C,
                                int N, int K){
  int tn = blockIdx.x, tm = blockIdx.y;
  int lane = threadIdx.x, half = lane >> 4, l16 = lane & 15;
  const bf16_t* arow0 = A + (size_t)(tm * 32 + l16) * K;
  const bf16_t* arow1 = A + (size_t)(tm * 32 + 16 + l16) * K;
  const bf16_t* wrow0 = W + (size_t)(tn * 64 + l16) * K;
  const bf16_t* wrow1 = W + (size_t)(tn * 64 + 16 + l16) * K;
  const bf16_t* wrow2 = W + (size_t)(tn * 64 + 32 + l16) * K;
  const bf16_t* wrow3 = W + (size_t)(tn * 64 + 48 + l16) * K;
  v8f c[2][4] = {};
  for (int k0 = 0; k0 < K; k0 += 32){
    __builtin_prefetch(arow0 + k0 + 256, 0, 0);
    __builtin_prefetch(arow1 + k0 + 256, 0, 0);
    v16bf av[2];
    {
      v8bf a0 = *(const v8bf*)(arow0 + k0 + half * 8);
      v8bf a1 = *(const v8bf*)(arow0 + k0 + 16 + half * 8);
      v8bf a2 = *(const v8bf*)(arow1 + k0 + half * 8);
      v8bf a3 = *(const v8bf*)(arow1 + k0 + 16 + half * 8);
#pragma unroll
      for (int i = 0; i < 8; i++){ av[0][i]=a0[i]; av[0][8+i]=a1[i]; av[1][i]=a2[i]; av[1][8+i]=a3[i]; }
    }
    const bf16_t* wr[4] = { wrow0, wrow1, wrow2, wrow3 };
#pragma unroll
    for (int j = 0; j < 4; j++){
      v8bf b0 = *(const v8bf*)(wr[j] + k0 + half * 8);
      v8bf b1 = *(const v8bf*)(wr[j] + k0 + 16 + half * 8);
      v16bf bv;
#pragma unroll
      for (int i = 0; i < 8; i++){ bv[i] = b0[i]; bv[8 + i] = b1[i]; }
      c[0][j] = __builtin_amdgcn_wmma_f32_16x16x32_bf16(false, av[0], false, bv, (short)0, c[0][j], false, false);
      c[1][j] = __builtin_amdgcn_wmma_f32_16x16x32_bf16(false, av[1], false, bv, (short)0, c[1][j], false, false);
    }
  }
#pragma unroll
  for (int j = 0; j < 4; j++){
    int col = tn * 64 + j * 16 + l16;
    float b = bias[col];
#pragma unroll
    for (int v = 0; v < 8; v++){
      C[(size_t)(tm * 32 + v + half * 8) * N + col]      = c[0][j][v] + b;
      C[(size_t)(tm * 32 + 16 + v + half * 8) * N + col] = c[1][j][v] + b;
    }
  }
}

// ---------------- RMSNorm rows ----------------
__global__ void k_rmsnorm(const float* __restrict__ X, const float* __restrict__ w,
                          float* __restrict__ Yf, bf16_t* __restrict__ Yb, int D){
  int row = blockIdx.x;
  const float* x = X + (size_t)row * D;
  __shared__ float red[8];
  float ss = 0.f;
  for (int i = threadIdx.x; i < D; i += blockDim.x){ float v = x[i]; ss += v * v; }
  for (int m = 16; m >= 1; m >>= 1) ss += __shfl_xor(ss, m, 32);
  if ((threadIdx.x & 31) == 0) red[threadIdx.x >> 5] = ss;
  __syncthreads();
  if (threadIdx.x == 0){
    float t = 0.f; int nw = blockDim.x >> 5;
    for (int i = 0; i < nw; i++) t += red[i];
    red[0] = t;
  }
  __syncthreads();
  float inv = rsqrtf(red[0] / (float)D + 1e-6f);
  for (int i = threadIdx.x; i < D; i += blockDim.x){
    float v = x[i] * inv * w[i];
    if (Yf) Yf[(size_t)row * D + i] = v;
    if (Yb) Yb[(size_t)row * D + i] = f2bf(v);
  }
}

// ---------------- RoPE cos/sin tables (RD=64 -> half=32) ----------------
__global__ void k_cossin(const int* __restrict__ start_pos,
                         float* __restrict__ cosb, float* __restrict__ sinb){
  int s = blockIdx.x; int i = threadIdx.x;
  float p = (float)(*start_pos + s);
  float inv = powf(10000.0f, -((float)i) / 32.0f);
  float ang = p * inv;
  cosb[s * 32 + i] = cosf(ang);
  sinb[s * 32 + i] = sinf(ang);
}

// ---------------- assemble q -> [H,S,192] bf16 ----------------
__global__ void k_assemble_q(const float* __restrict__ qn, const float* __restrict__ qr,
                             const float* __restrict__ cosb, const float* __restrict__ sinb,
                             bf16_t* __restrict__ qb){
  int s = blockIdx.x, h = blockIdx.y, d = threadIdx.x;
  float val;
  if (d < 128) val = qn[(size_t)s * 2048 + h * 128 + d];
  else {
    int j = d - 128;
    const float* r = qr + (size_t)s * 1024 + h * 64;
    if (j < 32) val = r[j] * cosb[s * 32 + j] - r[j + 32] * sinb[s * 32 + j];
    else { int i = j - 32; val = r[32 + i] * cosb[s * 32 + i] + r[i] * sinb[s * 32 + i]; }
  }
  qb[((size_t)h * S_LEN + s) * 192 + d] = f2bf(val);
}

// ---------------- assemble k -> [H,S,192] bf16 ----------------
__global__ void k_assemble_k(const float* __restrict__ kn, const float* __restrict__ kpe,
                             const float* __restrict__ cosb, const float* __restrict__ sinb,
                             bf16_t* __restrict__ kb){
  int s = blockIdx.x, h = blockIdx.y, d = threadIdx.x;
  float val;
  if (d < 128) val = kn[(size_t)s * 2048 + h * 128 + d];
  else {
    int j = d - 128;
    const float* r = kpe + (size_t)s * 64;
    if (j < 32) val = r[j] * cosb[s * 32 + j] - r[j + 32] * sinb[s * 32 + j];
    else { int i = j - 32; val = r[32 + i] * cosb[s * 32 + i] + r[i] * sinb[s * 32 + i]; }
  }
  kb[((size_t)h * S_LEN + s) * 192 + d] = f2bf(val);
}

// ---------------- assemble v^T -> [H,HD,S] bf16 ----------------
__global__ void k_assemble_vt(const float* __restrict__ v, bf16_t* __restrict__ vtb){
  int s = blockIdx.x, h = blockIdx.y, d = threadIdx.x;
  vtb[((size_t)h * 128 + d) * S_LEN + s] = f2bf(v[(size_t)s * 2048 + h * 128 + d]);
}

// ---------------- indexer row: concat/rope -> FWHT(128) -> fp8 blockwise quant-dequant ----------------
__global__ void k_indexer_row(const float* __restrict__ in, int h_per_s,
                              const float* __restrict__ cosb, const float* __restrict__ sinb,
                              bf16_t* __restrict__ out){
  int r = blockIdx.x;
  int s = r / h_per_s;
  int t = threadIdx.x;
  __shared__ float buf[128];
  const float* x = in + (size_t)r * 128;
  float val;
  if (t < 64) val = x[64 + t];
  else {
    int j = t - 64;
    if (j < 32) val = x[j] * cosb[s * 32 + j] - x[j + 32] * sinb[s * 32 + j];
    else { int i = j - 32; val = x[32 + i] * cosb[s * 32 + i] + x[i] * sinb[s * 32 + i]; }
  }
  buf[t] = val;
  __syncthreads();
  for (int hstep = 1; hstep < 128; hstep <<= 1){
    int i = 0; float a = 0.f, b = 0.f;
    if (t < 64){
      int grp = t / hstep, off = t % hstep;
      i = grp * (hstep * 2) + off;
      a = buf[i]; b = buf[i + hstep];
    }
    __syncthreads();
    if (t < 64){ buf[i] = a + b; buf[i + hstep] = a - b; }
    __syncthreads();
  }
  float xv = buf[t] * 0.08838834764831845f;
  float a = fabsf(xv);
  for (int m = 16; m >= 1; m >>= 1) a = fmaxf(a, __shfl_xor(a, m, 32));
  float sc = exp2f(ceilf(log2f(fmaxf(a / 448.0f, 1e-12f))));
  float q = fminf(fmaxf(xv / sc, -448.0f), 448.0f);
  out[(size_t)r * 128 + t] = f2bf(q * sc);
}

// ---------------- indexer score, 16x64 per wave (A reused across 4 t-tiles) ----------------
__global__ void k_score(const bf16_t* __restrict__ qdq, const bf16_t* __restrict__ kdq,
                        const float* __restrict__ widx, float* __restrict__ score){
  int tt = blockIdx.x, ts = blockIdx.y;
  int lane = threadIdx.x, half = lane >> 4, l16 = lane & 15;
  int srow = ts * 16 + l16;
  const bf16_t* brow[4];
#pragma unroll
  for (int j = 0; j < 4; j++) brow[j] = kdq + (size_t)(tt * 64 + j * 16 + l16) * 128;
  v8f acc[4] = {};
  for (int h = 0; h < 16; h++){
    v8f c[4] = {};
    const bf16_t* arow = qdq + (size_t)srow * 2048 + h * 128;
#pragma unroll
    for (int k0 = 0; k0 < 128; k0 += 32){
      v8bf a0 = *(const v8bf*)(arow + k0 + half * 8);
      v8bf a1 = *(const v8bf*)(arow + k0 + 16 + half * 8);
      v16bf av;
#pragma unroll
      for (int i = 0; i < 8; i++){ av[i] = a0[i]; av[8 + i] = a1[i]; }
#pragma unroll
      for (int j = 0; j < 4; j++){
        v8bf b0 = *(const v8bf*)(brow[j] + k0 + half * 8);
        v8bf b1 = *(const v8bf*)(brow[j] + k0 + 16 + half * 8);
        v16bf bv;
#pragma unroll
        for (int i = 0; i < 8; i++){ bv[i] = b0[i]; bv[8 + i] = b1[i]; }
        c[j] = __builtin_amdgcn_wmma_f32_16x16x32_bf16(false, av, false, bv, (short)0, c[j], false, false);
      }
    }
#pragma unroll
    for (int j = 0; j < 4; j++)
#pragma unroll
      for (int v = 0; v < 8; v++){
        int r = ts * 16 + v + half * 8;
        acc[j][v] += fmaxf(c[j][v], 0.f) * (widx[r * 16 + h] * 0.25f);
      }
  }
#pragma unroll
  for (int j = 0; j < 4; j++)
#pragma unroll
    for (int v = 0; v < 8; v++){
      int r = ts * 16 + v + half * 8;
      score[(size_t)r * S_LEN + tt * 64 + j * 16 + l16] = acc[j][v] * 0.08838834764831845f;
    }
}

// ---------------- exact top-512 per row via radix select; writes byte mask ----------------
__global__ void k_topk_mask(const float* __restrict__ score, unsigned char* __restrict__ mask){
  int row = blockIdx.x;
  const float* sr = score + (size_t)row * S_LEN;
  __shared__ unsigned hist[256];
  __shared__ unsigned scan[256];
  __shared__ unsigned prefix_s, rem_s;
  int t = threadIdx.x;
  unsigned u[8];
#pragma unroll
  for (int j = 0; j < 8; j++){
    unsigned b = __float_as_uint(sr[t * 8 + j]);
    u[j] = (b & 0x80000000u) ? ~b : (b | 0x80000000u);
  }
  unsigned prefix = 0, rem = TOPK_N;
  for (int pass = 3; pass >= 0; pass--){
    int shift = pass * 8;
    hist[t] = 0;
    __syncthreads();
    unsigned pm = (pass == 3) ? 0u : (0xFFFFFFFFu << ((pass + 1) * 8));
    for (int j = 0; j < 8; j++)
      if ((u[j] & pm) == prefix) atomicAdd(&hist[(u[j] >> shift) & 0xFF], 1u);
    __syncthreads();
    if (t == 0){
      unsigned c = 0; int b;
      for (b = 255; b >= 0; b--){
        if (c + hist[b] >= rem) break;
        c += hist[b];
      }
      if (b < 0) b = 0;
      prefix_s = prefix | ((unsigned)b << shift);
      rem_s = rem - c;
    }
    __syncthreads();
    prefix = prefix_s; rem = rem_s;
    __syncthreads();
  }
  unsigned thr = prefix, need = rem;
  unsigned eq = 0;
  for (int j = 0; j < 8; j++) eq += (u[j] == thr) ? 1u : 0u;
  scan[t] = eq; __syncthreads();
  for (int off = 1; off < 256; off <<= 1){
    unsigned v = (t >= off) ? scan[t - off] : 0u;
    __syncthreads();
    scan[t] += v;
    __syncthreads();
  }
  unsigned cnt = scan[t] - eq;
  unsigned char* mr = mask + (size_t)row * S_LEN;
  for (int j = 0; j < 8; j++){
    unsigned sel;
    if (u[j] > thr) sel = 1u;
    else if (u[j] == thr){ sel = (cnt < need) ? 1u : 0u; cnt++; }
    else sel = 0u;
    mr[t * 8 + j] = (unsigned char)sel;
  }
}

// ---------------- flash attention: one wave per (head, 16-row s-tile) ----------------
// K tiles double-buffered in LDS via CDNA5 async global->LDS loads (ASYNCcnt).
__global__ void k_attention(const bf16_t* __restrict__ qb, const bf16_t* __restrict__ kb,
                            const bf16_t* __restrict__ vtb, const unsigned char* __restrict__ mask,
                            float* __restrict__ o){
  int ts = blockIdx.x, h = blockIdx.y;
  int lane = threadIdx.x, half = lane >> 4, l16 = lane & 15;
  __shared__ __align__(16) bf16_t kbuf[2][32 * 192];  // 2 x 12KB
  __shared__ __align__(16) bf16_t pbuf[16 * 32];
  // Q (16x192) as six K=32 A-operands, kept in VGPRs
  v16bf qa[6];
  {
    const bf16_t* qrow = qb + ((size_t)h * S_LEN + ts * 16 + l16) * 192;
#pragma unroll
    for (int kc = 0; kc < 6; kc++){
      v8bf a0 = *(const v8bf*)(qrow + kc * 32 + half * 8);
      v8bf a1 = *(const v8bf*)(qrow + kc * 32 + 16 + half * 8);
#pragma unroll
      for (int i = 0; i < 8; i++){ qa[kc][i] = a0[i]; qa[kc][8 + i] = a1[i]; }
    }
  }
  v8f oacc[8] = {};
  float m[8], l[8];
#pragma unroll
  for (int v = 0; v < 8; v++){ m[v] = -1e30f; l[v] = 0.f; }
  const float sscale = 0.07216878364870322f;  // 192^-0.5

  // stage K tile 0 into kbuf[0]: 32 rows x 192 bf16 = 12288 contiguous bytes
  {
    unsigned lds = (unsigned)(uintptr_t)(&kbuf[0][0]) + (unsigned)(lane * 16);
    unsigned long long g = (unsigned long long)(uintptr_t)(kb + (size_t)h * S_LEN * 192) + (unsigned)(lane * 16);
    for (int j = 0; j < 24; j++){
      asm volatile("global_load_async_to_lds_b128 %0, %1, off" :: "v"(lds), "v"(g) : "memory");
      lds += 512; g += 512;
    }
  }

  for (int it = 0; it < S_LEN / 32; it++){
    int t0 = it * 32;
    int cur = it & 1;
    if (it + 1 < S_LEN / 32){
      WAIT_DS0();  // prior reads of the buffer we are about to overwrite are done
      unsigned lds = (unsigned)(uintptr_t)(&kbuf[cur ^ 1][0]) + (unsigned)(lane * 16);
      unsigned long long g = (unsigned long long)(uintptr_t)(kb + ((size_t)h * S_LEN + t0 + 32) * 192) + (unsigned)(lane * 16);
      for (int j = 0; j < 24; j++){
        asm volatile("global_load_async_to_lds_b128 %0, %1, off" :: "v"(lds), "v"(g) : "memory");
        lds += 512; g += 512;
      }
      WAIT_ASYNC24();  // current tile's 24 async loads complete; next tile in flight
    } else {
      WAIT_ASYNC0();
    }
    // scores for 16x32 strip from LDS-resident K tile
    v8f c0 = {}, c1 = {};
#pragma unroll
    for (int sub = 0; sub < 2; sub++){
      const bf16_t* krow = &kbuf[cur][(sub * 16 + l16) * 192];
#pragma unroll
      for (int kc = 0; kc < 6; kc++){
        v8bf b0 = *(const v8bf*)(krow + kc * 32 + half * 8);
        v8bf b1 = *(const v8bf*)(krow + kc * 32 + 16 + half * 8);
        v16bf bv;
#pragma unroll
        for (int i = 0; i < 8; i++){ bv[i] = b0[i]; bv[8 + i] = b1[i]; }
        if (sub == 0) c0 = __builtin_amdgcn_wmma_f32_16x16x32_bf16(false, qa[kc], false, bv, (short)0, c0, false, false);
        else          c1 = __builtin_amdgcn_wmma_f32_16x16x32_bf16(false, qa[kc], false, bv, (short)0, c1, false, false);
      }
    }
#pragma unroll
    for (int v = 0; v < 8; v++){
      int r = ts * 16 + v + half * 8;
      float x0 = c0[v] * sscale;
      float x1 = c1[v] * sscale;
      if (!mask[(size_t)r * S_LEN + t0 + l16])      x0 = -1e30f;
      if (!mask[(size_t)r * S_LEN + t0 + 16 + l16]) x1 = -1e30f;
      float mx = fmaxf(x0, x1);
      for (int sh = 8; sh >= 1; sh >>= 1) mx = fmaxf(mx, __shfl_xor(mx, sh, 32));
      float nm = fmaxf(m[v], mx);
      float f = expf(m[v] - nm);
      m[v] = nm;
      float p0 = expf(x0 - nm);
      float p1 = expf(x1 - nm);
      float rs = p0 + p1;
      for (int sh = 8; sh >= 1; sh >>= 1) rs += __shfl_xor(rs, sh, 32);
      l[v] = l[v] * f + rs;
#pragma unroll
      for (int i = 0; i < 8; i++) oacc[i][v] *= f;
      int rr = v + half * 8;
      pbuf[rr * 32 + l16]      = f2bf(p0);
      pbuf[rr * 32 + 16 + l16] = f2bf(p1);
    }
    __syncthreads();
    v16bf pa;
    {
      const bf16_t* prow = pbuf + l16 * 32;
      v8bf a0 = *(const v8bf*)(prow + half * 8);
      v8bf a1 = *(const v8bf*)(prow + 16 + half * 8);
#pragma unroll
      for (int i = 0; i < 8; i++){ pa[i] = a0[i]; pa[8 + i] = a1[i]; }
    }
#pragma unroll
    for (int dt = 0; dt < 8; dt++){
      const bf16_t* vrow = vtb + ((size_t)h * 128 + dt * 16 + l16) * S_LEN + t0;
      v8bf b0 = *(const v8bf*)(vrow + half * 8);
      v8bf b1 = *(const v8bf*)(vrow + 16 + half * 8);
      v16bf bv;
#pragma unroll
      for (int i = 0; i < 8; i++){ bv[i] = b0[i]; bv[8 + i] = b1[i]; }
      oacc[dt] = __builtin_amdgcn_wmma_f32_16x16x32_bf16(false, pa, false, bv, (short)0, oacc[dt], false, false);
    }
    __syncthreads();
  }
#pragma unroll
  for (int dt = 0; dt < 8; dt++)
#pragma unroll
    for (int v = 0; v < 8; v++){
      int r = ts * 16 + v + half * 8;
      o[(size_t)r * 2048 + h * 128 + dt * 16 + l16] = oacc[dt][v] / l[v];
    }
}

// =======================================================================
extern "C" void kernel_launch(void* const* d_in, const int* in_sizes, int n_in,
                              void* d_out, int out_size, void* d_ws, size_t ws_size,
                              hipStream_t stream) {
  const float* x          = (const float*)d_in[0];
  const int*   start_pos  = (const int*)d_in[1];
  const float* q_down_w   = (const float*)d_in[2];
  const float* q_down_b   = (const float*)d_in[3];
  const float* q_norm_w   = (const float*)d_in[4];
  const float* q_nope_w   = (const float*)d_in[5];
  const float* q_nope_b   = (const float*)d_in[6];
  const float* q_rope_w   = (const float*)d_in[7];
  const float* q_rope_b   = (const float*)d_in[8];
  const float* kv_down_w  = (const float*)d_in[9];
  const float* kv_down_b  = (const float*)d_in[10];
  const float* kv_norm_w  = (const float*)d_in[11];
  const float* k_up_w     = (const float*)d_in[12];
  const float* k_up_b     = (const float*)d_in[13];
  const float* v_up_w     = (const float*)d_in[14];
  const float* v_up_b     = (const float*)d_in[15];
  const float* k_rope_w   = (const float*)d_in[16];
  const float* k_rope_b   = (const float*)d_in[17];
  const float* out_w      = (const float*)d_in[18];
  const float* out_b      = (const float*)d_in[19];
  const float* idx_q_w    = (const float*)d_in[20];
  const float* idx_q_b    = (const float*)d_in[21];
  const float* idx_k_w    = (const float*)d_in[22];
  const float* idx_k_b    = (const float*)d_in[23];
  const float* idx_k_nw   = (const float*)d_in[24];
  const float* idx_w_w    = (const float*)d_in[25];
  const float* idx_w_b    = (const float*)d_in[26];

  char* ws = (char*)d_ws;
  size_t off = 0;
  auto alloc = [&](size_t bytes) -> void* {
    void* p = ws + off;
    off += (bytes + 255) & ~(size_t)255;
    return p;
  };

  const size_t S = S_LEN;
  bf16_t* xb   = (bf16_t*)alloc(S * DM * 2);
  bf16_t* wqd  = (bf16_t*)alloc((size_t)QLRN * DM * 2);
  bf16_t* wkvd = (bf16_t*)alloc((size_t)QLRN * DM * 2);
  bf16_t* wkr  = (bf16_t*)alloc((size_t)64 * DM * 2);
  bf16_t* wik  = (bf16_t*)alloc((size_t)128 * DM * 2);
  bf16_t* wiw  = (bf16_t*)alloc((size_t)16 * DM * 2);
  bf16_t* wqn  = (bf16_t*)alloc((size_t)2048 * 1024 * 2);
  bf16_t* wqr  = (bf16_t*)alloc((size_t)1024 * 1024 * 2);
  bf16_t* wku  = (bf16_t*)alloc((size_t)2048 * 1024 * 2);
  bf16_t* wvu  = (bf16_t*)alloc((size_t)2048 * 1024 * 2);
  bf16_t* wiq  = (bf16_t*)alloc((size_t)2048 * 1024 * 2);
  bf16_t* wow  = (bf16_t*)alloc((size_t)2048 * 2048 * 2);
  float*  tmp1 = (float*)alloc(S * 2048 * 4);
  bf16_t* qcb  = (bf16_t*)alloc(S * 1024 * 2);
  bf16_t* kvcb = (bf16_t*)alloc(S * 1024 * 2);
  float*  kpe  = (float*)alloc(S * 64 * 4);
  float*  cosb = (float*)alloc(S * 32 * 4);
  float*  sinb = (float*)alloc(S * 32 * 4);
  float*  qnope= (float*)alloc(S * 2048 * 4);
  float*  qrope= (float*)alloc(S * 1024 * 4);
  float*  knope= (float*)alloc(S * 2048 * 4);
  float*  vmat = (float*)alloc(S * 2048 * 4);
  bf16_t* qbm  = (bf16_t*)alloc((size_t)HN * S * 192 * 2);
  bf16_t* kbm  = (bf16_t*)alloc((size_t)HN * S * 192 * 2);
  bf16_t* vtb  = (bf16_t*)alloc((size_t)HN * 128 * S * 2);
  float*  qi   = (float*)alloc(S * 2048 * 4);
  float*  kif  = (float*)alloc(S * 128 * 4);
  bf16_t* qdq  = (bf16_t*)alloc(S * 2048 * 2);
  bf16_t* kdq  = (bf16_t*)alloc(S * 128 * 2);
  float*  widx = (float*)alloc(S * 16 * 4);
  float*  score= (float*)alloc(S * S * 4);
  unsigned char* maskb = (unsigned char*)alloc(S * S);
  float*  of   = (float*)alloc(S * 2048 * 4);
  bf16_t* ob   = (bf16_t*)alloc(S * 2048 * 2);

  auto cvt = [&](const float* src, bf16_t* dst, int n){
    int n4 = n / 4;
    k_cvt_bf16<<<(n4 + 255) / 256, 256, 0, stream>>>(src, dst, n4);
  };
  cvt(x, xb, S * DM);
  cvt(q_down_w, wqd, QLRN * DM);
  cvt(kv_down_w, wkvd, QLRN * DM);
  cvt(k_rope_w, wkr, 64 * DM);
  cvt(idx_k_w, wik, 128 * DM);
  cvt(idx_w_w, wiw, 16 * DM);
  cvt(q_nope_w, wqn, 2048 * 1024);
  cvt(q_rope_w, wqr, 1024 * 1024);
  cvt(k_up_w, wku, 2048 * 1024);
  cvt(v_up_w, wvu, 2048 * 1024);
  cvt(idx_q_w, wiq, 2048 * 1024);
  cvt(out_w, wow, 2048 * 2048);

  k_cossin<<<S, 32, 0, stream>>>(start_pos, cosb, sinb);

  auto gemm = [&](const bf16_t* A, const bf16_t* W, const float* bias, float* C, int N, int K){
    if ((N % 64) == 0)
      k_gemm_bf16_big<<<dim3(N / 64, S / 32), 32, 0, stream>>>(A, W, bias, C, N, K);
    else
      k_gemm_bf16<<<dim3(N / 16, S / 16), 32, 0, stream>>>(A, W, bias, C, N, K);
  };

  // q_c / kv_c
  gemm(xb, wqd, q_down_b, tmp1, 1024, 2048);
  k_rmsnorm<<<S, 256, 0, stream>>>(tmp1, q_norm_w, nullptr, qcb, 1024);
  gemm(xb, wkvd, kv_down_b, tmp1, 1024, 2048);
  k_rmsnorm<<<S, 256, 0, stream>>>(tmp1, kv_norm_w, nullptr, kvcb, 1024);
  // k_pe, ki, w_idx
  gemm(xb, wkr, k_rope_b, kpe, 64, 2048);
  gemm(xb, wik, idx_k_b, tmp1, 128, 2048);
  k_rmsnorm<<<S, 128, 0, stream>>>(tmp1, idx_k_nw, kif, nullptr, 128);
  gemm(xb, wiw, idx_w_b, widx, 16, 2048);
  // up-projections
  gemm(qcb, wqn, q_nope_b, qnope, 2048, 1024);
  gemm(qcb, wqr, q_rope_b, qrope, 1024, 1024);
  gemm(kvcb, wku, k_up_b, knope, 2048, 1024);
  gemm(kvcb, wvu, v_up_b, vmat, 2048, 1024);
  gemm(qcb, wiq, idx_q_b, qi, 2048, 1024);
  // assemble attention operands
  k_assemble_q<<<dim3(S, HN), 192, 0, stream>>>(qnope, qrope, cosb, sinb, qbm);
  k_assemble_k<<<dim3(S, HN), 192, 0, stream>>>(knope, kpe, cosb, sinb, kbm);
  k_assemble_vt<<<dim3(S, HN), 128, 0, stream>>>(vmat, vtb);
  // indexer
  k_indexer_row<<<S * HN, 128, 0, stream>>>(qi, HN, cosb, sinb, qdq);
  k_indexer_row<<<S, 128, 0, stream>>>(kif, 1, cosb, sinb, kdq);
  k_score<<<dim3(S / 64, S / 16), 32, 0, stream>>>(qdq, kdq, widx, score);
  k_topk_mask<<<S, 256, 0, stream>>>(score, maskb);
  // attention (async-LDS K staging)
  k_attention<<<dim3(S / 16, HN), 32, 0, stream>>>(qbm, kbm, vtb, maskb, of);
  // output projection
  cvt(of, ob, S * 2048);
  gemm(ob, wow, out_b, (float*)d_out, 2048, 2048);
  (void)in_sizes; (void)n_in; (void)out_size; (void)ws_size;
}
